// RCNN_10024453669695
// MI455X (gfx1250) — matrix operand order repaired and checked by reading, not compiled, not run on hardware
//
#include <hip/hip_runtime.h>

// ---------------------------------------------------------------------------
// Types for CDNA5 WMMA (wave32): v_wmma_f32_16x16x32_bf16
// ---------------------------------------------------------------------------
typedef __bf16 bf16;
typedef __attribute__((ext_vector_type(8)))  bf16  v8bf;
typedef __attribute__((ext_vector_type(16))) bf16  v16bf;
typedef __attribute__((ext_vector_type(8)))  float v8f;

// Problem constants
#define SEQ   512
#define BATCH 64
#define EMB   512
#define HID   512
#define MAXD  1024
#define NCLS  20
#define CATW  1536            // 2*HID + EMB
#define ROWS  (SEQ * BATCH)   // 32768

// Workspace layout (bytes)
#define CAT_OFF   0u                          // bf16 [ROWS][1536]  (c_ls|emb|c_rs)
#define ELS_OFF   100663296u                  // bf16 [ROWS][512]
#define ERS_OFF   134217728u
#define WSL_OFF   167772160u                  // bf16 512x512
#define WSR_OFF   168296448u
#define WL_OFF    168820736u
#define WR_OFF    169345024u
#define WMAX_OFF  169869312u                  // bf16 1024x1536
#define YMAX_OFF  173015040u                  // f32  [64][1024]

__device__ __forceinline__ v8f wmma_bf16(v16bf a, v16bf b, v8f c) {
  return __builtin_amdgcn_wmma_f32_16x16x32_bf16(
      /*neg_a=*/false, a, /*neg_b=*/false, b,
      /*c_mod=*/(short)0, c, /*reuse_a=*/false, /*reuse_b=*/false);
}

// A fragment (16x32, MxK): lane<16 row=lane holds K{kb..kb+7, kb+16..kb+23};
// lane>=16 row=lane-16 holds K{kb+8..kb+15, kb+24..kb+31}. rowbase = &A[row][0].
__device__ __forceinline__ v16bf load_a_frag(const bf16* rowbase, int kbase, int lane) {
  union { v16bf v; v8bf h[2]; } u;
  const bf16* p = rowbase + kbase + ((lane & 16) ? 8 : 0);
  u.h[0] = *(const v8bf*)(p);
  u.h[1] = *(const v8bf*)(p + 16);
  return u.v;
}

// B fragment (32x16, KxN): lane<16 col=lane holds K{kb..kb+7, kb+16..kb+23};
// lane>=16 col=lane-16 holds K{kb+8..kb+15, kb+24..kb+31}. colbase = &W[col][0] (N x K row-major).
__device__ __forceinline__ v16bf load_b_frag(const bf16* colbase, int kbase, int lane) {
  union { v16bf v; v8bf h[2]; } u;
  const bf16* p = colbase + kbase + ((lane & 16) ? 8 : 0);
  u.h[0] = *(const v8bf*)(p);
  u.h[1] = *(const v8bf*)(p + 16);
  return u.v;
}

// ---------------------------------------------------------------------------
// K0: fp32 -> bf16 weight conversion
// ---------------------------------------------------------------------------
__global__ __launch_bounds__(256) void cvt_bf16(const float* __restrict__ src,
                                                bf16* __restrict__ dst, int n) {
  int i = blockIdx.x * 256 + threadIdx.x;
  if (i < n) dst[i] = (bf16)src[i];
}

// ---------------------------------------------------------------------------
// K1: embedding gather -> cat[:, 512:1024] (bf16)
// ---------------------------------------------------------------------------
__global__ __launch_bounds__(256) void gather_emb(const int* __restrict__ inp,
                                                  const float* __restrict__ tab,
                                                  bf16* __restrict__ cat) {
  const size_t total = (size_t)ROWS * EMB;
  size_t stride = (size_t)gridDim.x * 256;
  for (size_t idx = (size_t)blockIdx.x * 256 + threadIdx.x; idx < total; idx += stride) {
    int row = (int)(idx >> 9);
    int e   = (int)(idx & 511);
    int tok = inp[row];
    cat[(size_t)row * CATW + HID + e] = (bf16)tab[(size_t)tok * EMB + e];
  }
}

// ---------------------------------------------------------------------------
// K2: emb_ls = emb @ W_sl^T + b_sl ; emb_rs = emb @ W_sr^T + b_sr  (bf16 out)
// Register-blocked: each wave computes a 16(M) x 64(N) strip: one A-frag feeds
// 4 B-frags / 4 accumulators per K-step (4x less A traffic from L2).
// Jobs = 2 matrices x 2048 mt x 8 nt4 = 32768 waves -> 4096 blocks x 8 waves.
// ---------------------------------------------------------------------------
__global__ __launch_bounds__(256) void gemm_se(const bf16* __restrict__ cat,
                                               const bf16* __restrict__ Wsl,
                                               const bf16* __restrict__ Wsr,
                                               const float* __restrict__ bsl,
                                               const float* __restrict__ bsr,
                                               bf16* __restrict__ els,
                                               bf16* __restrict__ ers) {
  int lane  = threadIdx.x & 31;
  int gw    = blockIdx.x * 8 + (threadIdx.x >> 5);   // 0..32767
  int which = gw >> 14;                               // 0: W_sl, 1: W_sr
  int rem   = gw & 16383;
  int mt    = rem >> 3;                               // 0..2047
  int nt4   = rem & 7;                                // 0..7 (N strip of 64)

  const bf16*  W    = which ? Wsr : Wsl;
  const float* bias = which ? bsr : bsl;
  bf16*        outp = which ? ers : els;

  const bf16* arow = cat + (size_t)(mt * 16 + (lane & 15)) * CATW + HID; // emb region
  const bf16* bcol0 = W + (size_t)(nt4 * 64 + (lane & 15)) * EMB;

  v8f acc[4] = {{}, {}, {}, {}};
  for (int kk = 0; kk < 16; ++kk) {
    v16bf a = load_a_frag(arow, kk * 32, lane);
#pragma unroll
    for (int sub = 0; sub < 4; ++sub) {
      v16bf b = load_b_frag(bcol0 + (size_t)sub * 16 * EMB, kk * 32, lane);
      acc[sub] = wmma_bf16(a, b, acc[sub]);
    }
  }
  int mbase = mt * 16 + ((lane & 16) ? 8 : 0);
#pragma unroll
  for (int sub = 0; sub < 4; ++sub) {
    int   h  = nt4 * 64 + sub * 16 + (lane & 15);
    float bh = bias[h];
    for (int i = 0; i < 8; ++i)
      outp[(size_t)(mbase + i) * HID + h] = (bf16)(acc[sub][i] + bh);
  }
}

// ---------------------------------------------------------------------------
// K3: sequential scans. grid=2 (blockIdx.x: 0=fwd left, 1=bwd right),
// 1024 threads = 32 waves. c state (64x512 bf16 = 64 KB) in LDS.
// Per step: 4x32 = 128 output tiles of 16x16, 4 tiles per wave, K=512.
// New state staged in registers across a barrier (no LDS double buffer).
// ---------------------------------------------------------------------------
__global__ __launch_bounds__(1024) void rnn_scan(const bf16* __restrict__ Wl,
                                                 const bf16* __restrict__ Wr,
                                                 const float* __restrict__ bl,
                                                 const float* __restrict__ br,
                                                 const float* __restrict__ cl0,
                                                 const float* __restrict__ cr0,
                                                 const bf16* __restrict__ els,
                                                 const bf16* __restrict__ ers,
                                                 bf16* __restrict__ cat) {
  __shared__ bf16 c_lds[BATCH * HID];   // 64 KB

  const int dir = blockIdx.x;
  const bf16*  W    = dir ? Wr  : Wl;
  const float* bias = dir ? br  : bl;
  const float* c0   = dir ? cr0 : cl0;
  const bf16*  e    = dir ? ers : els;
  const int    coff = dir ? (HID + EMB) : 0;

  int tid = threadIdx.x, lane = tid & 31, wave = tid >> 5;

  for (int i = tid; i < BATCH * HID; i += 1024)
    c_lds[i] = (bf16)c0[i & (HID - 1)];
  __syncthreads();

  for (int step = 0; step < SEQ; ++step) {
    int s = dir ? (SEQ - 1 - step) : step;
    float out[4][8];

    for (int t = 0; t < 4; ++t) {
      int id = wave + t * 32;            // 0..127
      int mt = id >> 5;                  // 0..3
      int nt = id & 31;                  // 0..31
      const bf16* arow = c_lds + (size_t)(mt * 16 + (lane & 15)) * HID;
      const bf16* bcol = W + (size_t)(nt * 16 + (lane & 15)) * HID;
      v8f acc = {};
      for (int kk = 0; kk < 16; ++kk) {
        v16bf a = load_a_frag(arow, kk * 32, lane);
        v16bf b = load_b_frag(bcol, kk * 32, lane);
        acc = wmma_bf16(a, b, acc);
      }
      int   h     = nt * 16 + (lane & 15);
      float bh    = bias[h];
      int   mbase = mt * 16 + ((lane & 16) ? 8 : 0);
      for (int i = 0; i < 8; ++i) {
        int   brow = mbase + i;
        float ev   = (float)e[(size_t)(s * BATCH + brow) * HID + h];
        out[t][i]  = tanhf(acc[i] + bh + ev);
      }
    }
    __syncthreads();                     // all reads of c done

    for (int t = 0; t < 4; ++t) {
      int id = wave + t * 32;
      int mt = id >> 5, nt = id & 31;
      int h     = nt * 16 + (lane & 15);
      int mbase = mt * 16 + ((lane & 16) ? 8 : 0);
      for (int i = 0; i < 8; ++i) {
        int  brow = mbase + i;
        bf16 v    = (bf16)out[t][i];
        c_lds[brow * HID + h] = v;
        cat[(size_t)(s * BATCH + brow) * CATW + coff + h] = v;
      }
    }
    __syncthreads();
  }
}

// ---------------------------------------------------------------------------
// K4: y = tanh(cat @ W_max^T + b_max); y_max = max over s.
// Grid = 4 (b-tiles) x 64 (m-tiles) = 256 blocks x 256 thr (8 waves).
// W_max tile (16 x 1536) staged in LDS (padded stride, conflict-free).
// Each wave processes TWO s-rows per iteration sharing each B-frag (halves
// LDS traffic), keeps a running per-lane max, LDS cross-wave max-reduce,
// single write per output element (no atomics). cat (100 MB) is L2-resident.
// ---------------------------------------------------------------------------
__global__ __launch_bounds__(256) void gemm_max(const bf16* __restrict__ cat,
                                                const bf16* __restrict__ Wmax,
                                                const float* __restrict__ bmax,
                                                float* __restrict__ ymax) {
  __shared__ bf16  wlds[16 * 1544];     // padded row stride avoids bank conflicts
  __shared__ float red[8 * 256];

  int bt = blockIdx.x >> 6;             // 0..3
  int nt = blockIdx.x & 63;             // 0..63
  int tid = threadIdx.x, lane = tid & 31, wave = tid >> 5;

  for (int i = tid; i < 16 * CATW; i += 256) {
    int hl = i / CATW, k = i % CATW;
    wlds[hl * 1544 + k] = Wmax[(size_t)(nt * 16 + hl) * CATW + k];
  }
  __syncthreads();

  int   n  = lane & 15;
  float bh = bmax[nt * 16 + n];
  const bf16* bcol = &wlds[n * 1544];

  v8f vmax;
  for (int i = 0; i < 8; ++i) vmax[i] = -3.0e38f;

  for (int s = wave * 2; s < SEQ; s += 16) {   // 32 iters x 2 s-rows = 64 s/wave
    const bf16* arow0 = cat + (size_t)((s)     * BATCH + bt * 16 + (lane & 15)) * CATW;
    const bf16* arow1 = cat + (size_t)((s + 1) * BATCH + bt * 16 + (lane & 15)) * CATW;
    v8f acc0 = {}, acc1 = {};
    for (int kk = 0; kk < 48; ++kk) {
      v16bf b  = load_b_frag(bcol, kk * 32, lane);
      v16bf a0 = load_a_frag(arow0, kk * 32, lane);
      v16bf a1 = load_a_frag(arow1, kk * 32, lane);
      acc0 = wmma_bf16(a0, b, acc0);
      acc1 = wmma_bf16(a1, b, acc1);
    }
    for (int i = 0; i < 8; ++i) {
      vmax[i] = fmaxf(vmax[i], tanhf(acc0[i] + bh));
      vmax[i] = fmaxf(vmax[i], tanhf(acc1[i] + bh));
    }
  }

  int mbase = (lane & 16) ? 8 : 0;
  for (int i = 0; i < 8; ++i)
    red[wave * 256 + (mbase + i) * 16 + n] = vmax[i];
  __syncthreads();

  float m = red[tid];
  for (int w = 1; w < 8; ++w) m = fmaxf(m, red[w * 256 + tid]);
  int mm = tid >> 4, nn = tid & 15;
  ymax[(size_t)(bt * 16 + mm) * MAXD + nt * 16 + nn] = m;
}

// ---------------------------------------------------------------------------
// K5: logits = y_max @ W_doc^T + b_doc; log_softmax. fp32, 64 threads (1/b).
// ---------------------------------------------------------------------------
__global__ __launch_bounds__(64) void doc_head(const float* __restrict__ ymax,
                                               const float* __restrict__ Wdoc,
                                               const float* __restrict__ bdoc,
                                               float* __restrict__ outp) {
  int b = threadIdx.x;
  const float* yr = ymax + (size_t)b * MAXD;
  float logits[NCLS];
  for (int k = 0; k < NCLS; ++k) {
    float acc = bdoc[k];
    const float* wr = Wdoc + (size_t)k * MAXD;
    for (int m = 0; m < MAXD; ++m) acc += yr[m] * wr[m];
    logits[k] = acc;
  }
  float mx = logits[0];
  for (int k = 1; k < NCLS; ++k) mx = fmaxf(mx, logits[k]);
  float se = 0.f;
  for (int k = 0; k < NCLS; ++k) se += expf(logits[k] - mx);
  float lse = mx + logf(se);
  for (int k = 0; k < NCLS; ++k) outp[b * NCLS + k] = logits[k] - lse;
}

// ---------------------------------------------------------------------------
extern "C" void kernel_launch(void* const* d_in, const int* in_sizes, int n_in,
                              void* d_out, int out_size, void* d_ws, size_t ws_size,
                              hipStream_t stream) {
  const int*   inp    = (const int*)d_in[0];
  const float* table  = (const float*)d_in[1];
  const float* c_l0   = (const float*)d_in[2];
  const float* c_r0   = (const float*)d_in[3];
  const float* W_l    = (const float*)d_in[4];
  const float* b_l    = (const float*)d_in[5];
  const float* W_r    = (const float*)d_in[6];
  const float* b_r    = (const float*)d_in[7];
  const float* W_sl   = (const float*)d_in[8];
  const float* b_sl   = (const float*)d_in[9];
  const float* W_sr   = (const float*)d_in[10];
  const float* b_sr   = (const float*)d_in[11];
  const float* W_max  = (const float*)d_in[12];
  const float* b_max  = (const float*)d_in[13];
  const float* W_doc  = (const float*)d_in[14];
  const float* b_doc  = (const float*)d_in[15];

  char* ws = (char*)d_ws;
  bf16*  cat   = (bf16*)(ws + CAT_OFF);
  bf16*  els   = (bf16*)(ws + ELS_OFF);
  bf16*  ers   = (bf16*)(ws + ERS_OFF);
  bf16*  wsl   = (bf16*)(ws + WSL_OFF);
  bf16*  wsr   = (bf16*)(ws + WSR_OFF);
  bf16*  wl    = (bf16*)(ws + WL_OFF);
  bf16*  wr    = (bf16*)(ws + WR_OFF);
  bf16*  wmax  = (bf16*)(ws + WMAX_OFF);
  float* ymax  = (float*)(ws + YMAX_OFF);

  // Weight conversions (fp32 -> bf16)
  cvt_bf16<<<(HID * HID + 255) / 256, 256, 0, stream>>>(W_sl, wsl, HID * HID);
  cvt_bf16<<<(HID * HID + 255) / 256, 256, 0, stream>>>(W_sr, wsr, HID * HID);
  cvt_bf16<<<(HID * HID + 255) / 256, 256, 0, stream>>>(W_l, wl, HID * HID);
  cvt_bf16<<<(HID * HID + 255) / 256, 256, 0, stream>>>(W_r, wr, HID * HID);
  cvt_bf16<<<(MAXD * CATW + 255) / 256, 256, 0, stream>>>(W_max, wmax, MAXD * CATW);

  // Embedding gather into cat[:, 512:1024]
  gather_emb<<<16384, 256, 0, stream>>>(inp, table, cat);

  // emb_ls / emb_rs (register-blocked 16x64 per wave)
  gemm_se<<<4096, 256, 0, stream>>>(cat, wsl, wsr, b_sl, b_sr, els, ers);

  // Bidirectional recurrent scans -> cat[:, 0:512] and cat[:, 1024:1536]
  rnn_scan<<<2, 1024, 0, stream>>>(wl, wr, b_l, b_r, c_l0, c_r0, els, ers, cat);

  // y = tanh(cat @ W_max^T + b_max), max over s
  gemm_max<<<256, 256, 0, stream>>>(cat, wmax, b_max, ymax);

  // doc logits + log_softmax
  doc_head<<<1, 64, 0, stream>>>(ymax, W_doc, b_doc, (float*)d_out);
}
